// Moe_GCN_Transformer_unit_89541478187746
// MI455X (gfx1250) — matrix operand order repaired and checked by reading, not compiled, not run on hardware
//
#include <hip/hip_runtime.h>

// ---------------- problem constants (from reference) ----------------
#define NB    4
#define TT    2048
#define NT    (NB * TT)          // 8192 rows
#define HH    8
#define DH    64
#define VV    5
#define CC    64
#define EE    8
#define DIM   320                // V*C
#define INNER 512                // H*DH
#define TH3   1536               // 3*INNER
#define SCALE 0.125f             // DH^-0.5

typedef __attribute__((ext_vector_type(16))) __bf16    v16bf;
typedef __attribute__((ext_vector_type(8)))  float     v8f;
typedef __attribute__((ext_vector_type(4)))  unsigned  v4u;
typedef __attribute__((ext_vector_type(8)))  unsigned  v8u;

// ---------------- f32 -> bf16 conversion ----------------
__global__ void cvt_bf16_kernel(const float* __restrict__ src, __bf16* __restrict__ dst, int n) {
  int i = blockIdx.x * blockDim.x + threadIdx.x;
  if (i < n) dst[i] = (__bf16)src[i];
}

// ---------------- generic tiled WMMA bf16 GEMM ----------------
// C[M,N] = A[M,K] @ B[K,N]; A,B row-major bf16; 256 threads = 8 waves,
// each wave computes one 16x16 C tile; workgroup covers 64x32 of C.
template <int OUT_F32>
__global__ void gemm_bf16_kernel(const __bf16* __restrict__ A, const __bf16* __restrict__ B,
                                 float* __restrict__ Cf, __bf16* __restrict__ Cb,
                                 int M, int N, int K) {
  const int w    = threadIdx.x >> 5;
  const int lane = threadIdx.x & 31;
  const int mt = w >> 1, nt = w & 1;
  const int mbase = blockIdx.x * 64 + mt * 16;
  const int nbase = blockIdx.y * 32 + nt * 16;
  const int ml = lane & 15;      // A row within tile / B column within tile
  const int hi = lane >> 4;      // lane half selects K sub-range

  const __bf16* Arow = A + (size_t)(mbase + ml) * K;
  v8f c = {};
  for (int k0 = 0; k0 < K; k0 += 32) {
    v16bf a, b;
    const int kba = k0 + hi * 16;  // A half-lane K base (contiguous K, matches B)
    #pragma unroll
    for (int j = 0; j < 16; ++j) a[j] = Arow[kba + j];
    const int kbb = k0 + hi * 16;  // B half-lane K base (contiguous K, matches A)
    #pragma unroll
    for (int j = 0; j < 16; ++j) b[j] = B[(size_t)(kbb + j) * N + nbase + ml];
    c = __builtin_amdgcn_wmma_f32_16x16x32_bf16(false, a, false, b, (short)0, c, false, false);
  }
  #pragma unroll
  for (int r = 0; r < 8; ++r) {
    size_t off = (size_t)(mbase + r + hi * 8) * N + nbase + ml;
    if (OUT_F32) Cf[off] = c[r];
    else         Cb[off] = (__bf16)c[r];
  }
}

// ---------------- fused attention: softmax(QK^T * s) @ V ----------------
// grid = (T/16, H, N), block = 256 (8 waves). LDS row-block of scores:
//   S   : 16 x 2048 f32   (131072 B)
//   P   : 16 x 2048 bf16  ( 65536 B)  unnormalized exp
//   Qs  : 16 x 64  bf16   (  2048 B)  staged by the Tensor Data Mover
//   red : 16 x 16  f32, rowmax/rowsum, Opart: 8 x 16 x 16 f32
__global__ void attention_kernel(const __bf16* __restrict__ qkv, __bf16* __restrict__ out) {
  extern __shared__ char smem[];
  float*  S      = (float*)smem;                       // 131072
  __bf16* P      = (__bf16*)(smem + 131072);           //  65536
  __bf16* Qs     = (__bf16*)(smem + 196608);           //   2048
  float*  red    = (float*)(smem + 198656);            //   1024
  float*  rowmax = (float*)(smem + 199680);            //     64
  float*  rowsum = (float*)(smem + 199744);            //     64
  float*  Opart  = (float*)(smem + 199808);            //   8192  -> 208000 total

  const int t0 = blockIdx.x * 16;
  const int h  = blockIdx.y;
  const int n  = blockIdx.z;
  const int tid  = threadIdx.x;
  const int w    = tid >> 5;
  const int lane = tid & 31;
  const int ml = lane & 15;
  const int hi = lane >> 4;

  // --- TDM: DMA the 16x64 bf16 Q tile (row stride TH3 elements) into LDS ---
  // D# per CDNA5 ISA; issued once (wave 0), EXEC ignored.
  if (w == 0) {
    const unsigned long long gaddr =
        (unsigned long long)(uintptr_t)(qkv + (size_t)(n * TT + t0) * TH3 + h * DH);
    const unsigned laddr = (unsigned)(uintptr_t)Qs;   // LDS byte offset
    v4u g0;
    g0[0] = 1u;                                       // count=1 valid user D#
    g0[1] = laddr;                                    // lds_addr
    g0[2] = (unsigned)gaddr;                          // global_addr[31:0]
    g0[3] = (unsigned)(gaddr >> 32) | 0x80000000u;    // global_addr[56:32] | type=2
    v8u g1;
    g1[0] = 0x00010000u;   // workgroup_mask=0, data_size=1 (2-byte elements)
    g1[1] = 64u << 16;     // tensor_dim0 = 64 (low 16 bits at [31:16])
    g1[2] = 16u << 16;     // tensor_dim0 hi = 0, tensor_dim1 = 16
    g1[3] = 64u << 16;     // tensor_dim1 hi = 0, tile_dim0 = 64
    g1[4] = 16u;           // tile_dim1 = 16, tile_dim2 = 0
    g1[5] = (unsigned)TH3; // tensor_dim0_stride[31:0] = 1536
    g1[6] = 0u;            // stride0 hi = 0, tensor_dim1_stride lo = 0 (2D tile)
    g1[7] = 0u;
    asm volatile("tensor_load_to_lds %0, %1" :: "s"(g0), "s"(g1) : "memory");
    __builtin_amdgcn_s_wait_tensorcnt(0);
  }
  __syncthreads();

  // per-wave Q fragments for the two K=32 steps
  v16bf aq[2];
  #pragma unroll
  for (int s = 0; s < 2; ++s) {
    const int kb = s * 32 + hi * 16;
    #pragma unroll
    for (int j = 0; j < 16; ++j) aq[s][j] = Qs[ml * 64 + kb + j];
  }

  // scores: each wave sweeps key tiles jt = w, w+8, ...
  for (int jt = w; jt < TT / 16; jt += 8) {
    const int jbase = jt * 16;
    const __bf16* Kp = qkv + (size_t)(n * TT + jbase + ml) * TH3 + INNER + h * DH;
    v8f c = {};
    #pragma unroll
    for (int s = 0; s < 2; ++s) {
      v16bf b;
      const int kb = s * 32 + hi * 16;
      #pragma unroll
      for (int j = 0; j < 16; ++j) b[j] = Kp[kb + j];
      c = __builtin_amdgcn_wmma_f32_16x16x32_bf16(false, aq[s], false, b, (short)0, c, false, false);
    }
    #pragma unroll
    for (int r = 0; r < 8; ++r) S[(r + hi * 8) * TT + jbase + ml] = c[r];
  }
  __syncthreads();

  // softmax over each of the 16 rows; 16 threads per row
  {
    const int row = tid >> 4, sub = tid & 15;
    float mx = -3.0e38f;
    for (int j = sub; j < TT; j += 16) mx = fmaxf(mx, S[row * TT + j]);
    red[row * 16 + sub] = mx;
    __syncthreads();
    if (sub == 0) {
      float m = red[row * 16];
      for (int i = 1; i < 16; ++i) m = fmaxf(m, red[row * 16 + i]);
      rowmax[row] = m * SCALE;
    }
    __syncthreads();
    const float rm = rowmax[row];
    float sm = 0.f;
    for (int j = sub; j < TT; j += 16) {
      float p = __expf(S[row * TT + j] * SCALE - rm);
      P[row * TT + j] = (__bf16)p;
      sm += p;
    }
    red[row * 16 + sub] = sm;
    __syncthreads();
    if (sub == 0) {
      float s = 0.f;
      for (int i = 0; i < 16; ++i) s += red[row * 16 + i];
      rowsum[row] = s;
    }
    __syncthreads();
  }

  // PV: wave w -> d-tile (w&3), K-half (w>>2); O[16,64] = P[16,2048] @ V[2048,64]
  {
    const int dt = w & 3, half = w >> 2;
    const int kstart = half * 1024;
    const __bf16* Vcol = qkv + 2 * INNER + h * DH + dt * 16 + ml;
    v8f o = {};
    for (int ks = 0; ks < 1024; ks += 32) {
      const int k0 = kstart + ks;
      v16bf a, b;
      const int kba = k0 + hi * 16;
      #pragma unroll
      for (int j = 0; j < 16; ++j) a[j] = P[ml * TT + kba + j];
      const int kbb = k0 + hi * 16;
      #pragma unroll
      for (int j = 0; j < 16; ++j) b[j] = Vcol[(size_t)(n * TT + kbb + j) * TH3];
      o = __builtin_amdgcn_wmma_f32_16x16x32_bf16(false, a, false, b, (short)0, o, false, false);
    }
    #pragma unroll
    for (int r = 0; r < 8; ++r) Opart[w * 256 + (r + hi * 8) * 16 + ml] = o[r];
  }
  __syncthreads();

  // combine K-halves, normalize by row sums, emit bf16 attention output
  for (int i = tid; i < 16 * 64; i += 256) {
    const int mrow = i >> 6, d = i & 63;
    const int tile = d >> 4, dn = d & 15;
    float val = Opart[tile * 256 + mrow * 16 + dn] + Opart[(4 + tile) * 256 + mrow * 16 + dn];
    val /= rowsum[mrow];
    out[(size_t)(n * TT + t0 + mrow) * INNER + h * DH + d] = (__bf16)val;
  }
}

// ---------------- LN1 + LeakyReLU + residual ----------------
__global__ void ln1_kernel(const float* __restrict__ oproj, const float* __restrict__ bo,
                           const float* __restrict__ g, const float* __restrict__ b,
                           const float* __restrict__ x, float* __restrict__ y) {
  __shared__ float sh[512];
  __shared__ float stat[2];
  const int t = threadIdx.x;
  const size_t row = blockIdx.x;
  float val = oproj[row * DIM + t] + bo[t];
  sh[t] = val;
  if (t < 192) sh[320 + t] = 0.f;
  __syncthreads();
  for (int s = 256; s > 0; s >>= 1) { if (t < s) sh[t] += sh[t + s]; __syncthreads(); }
  if (t == 0) stat[0] = sh[0] * (1.f / 320.f);
  __syncthreads();
  const float dv = val - stat[0];
  sh[t] = dv * dv;
  if (t < 192) sh[320 + t] = 0.f;
  __syncthreads();
  for (int s = 256; s > 0; s >>= 1) { if (t < s) sh[t] += sh[t + s]; __syncthreads(); }
  if (t == 0) stat[1] = sh[0] * (1.f / 320.f);
  __syncthreads();
  float nv = dv * rsqrtf(stat[1] + 1e-5f) * g[t] + b[t];
  float lr = nv > 0.f ? nv : 0.01f * nv;
  y[row * DIM + t] = lr + x[row * DIM + t];
}

// ---------------- MoE gate (argmax over E) + masked per-expert GCN ----------------
__global__ void moe_gcn_kernel(const float* __restrict__ y, const float* __restrict__ A,
                               const float* __restrict__ Wg, const float* __restrict__ bg,
                               const float* __restrict__ Wgcn, const float* __restrict__ bgcn,
                               float* __restrict__ yout) {
  __shared__ float ys[DIM];
  __shared__ float dinv[VV];
  __shared__ float nA[VV * VV];
  __shared__ int   idx[CC];
  const int t = threadIdx.x;
  const size_t row = blockIdx.x;
  ys[t] = y[row * DIM + t];
  if (t < VV) {
    float s = 0.f;
    for (int j = 0; j < VV; ++j) s += A[t * VV + j];
    dinv[t] = rsqrtf(s);
  }
  __syncthreads();
  if (t < VV * VV) nA[t] = A[t] * dinv[t / VV] * dinv[t % VV];
  if (t < CC) {
    const int c = t;
    float best = -3.0e38f; int bi = 0;
    for (int e = 0; e < EE; ++e) {
      float p = bg[e];
      for (int v = 0; v < VV; ++v) p += ys[v * CC + c] * Wg[v * EE + e];
      if (p > best) { best = p; bi = e; }
    }
    idx[c] = bi;
  }
  __syncthreads();
  const int v = t >> 6, d = t & 63;
  const int e = idx[d];
  const float* We = Wgcn + (size_t)e * CC * CC;
  float acc = bgcn[e * CC + d];
  for (int vp = 0; vp < VV; ++vp) {
    const float na = nA[v * VV + vp];
    const float* yrow = &ys[vp * CC];
    for (int c = 0; c < CC; ++c)
      if (idx[c] == e) acc += na * yrow[c] * We[c * CC + d];
  }
  yout[row * DIM + t] = ys[t] + acc;
}

// ---------------- causal 1x3 conv over time (left pad 2) ----------------
__global__ void conv_kernel(const float* __restrict__ yin, const float* __restrict__ cw,
                            const float* __restrict__ cb, float* __restrict__ yconv) {
  const int i = blockIdx.x * blockDim.x + threadIdx.x;
  if (i >= NT * DIM) return;
  const int t = (i / DIM) % TT;
  float v = cw[2] * yin[i] + cb[0];
  if (t >= 1) v += cw[1] * yin[i - DIM];
  if (t >= 2) v += cw[0] * yin[i - 2 * DIM];
  yconv[i] = v;
}

// ---------------- LN2 + LeakyReLU -> output ----------------
__global__ void ln2_kernel(const float* __restrict__ yconv, const float* __restrict__ g,
                           const float* __restrict__ b, float* __restrict__ out) {
  __shared__ float sh[512];
  __shared__ float stat[2];
  const int t = threadIdx.x;
  const size_t row = blockIdx.x;
  float val = yconv[row * DIM + t];
  sh[t] = val;
  if (t < 192) sh[320 + t] = 0.f;
  __syncthreads();
  for (int s = 256; s > 0; s >>= 1) { if (t < s) sh[t] += sh[t + s]; __syncthreads(); }
  if (t == 0) stat[0] = sh[0] * (1.f / 320.f);
  __syncthreads();
  const float dv = val - stat[0];
  sh[t] = dv * dv;
  if (t < 192) sh[320 + t] = 0.f;
  __syncthreads();
  for (int s = 256; s > 0; s >>= 1) { if (t < s) sh[t] += sh[t + s]; __syncthreads(); }
  if (t == 0) stat[1] = sh[0] * (1.f / 320.f);
  __syncthreads();
  float nv = dv * rsqrtf(stat[1] + 1e-5f) * g[t] + b[t];
  out[row * DIM + t] = nv > 0.f ? nv : 0.01f * nv;
}

// ---------------- launcher ----------------
extern "C" void kernel_launch(void* const* d_in, const int* in_sizes, int n_in,
                              void* d_out, int out_size, void* d_ws, size_t ws_size,
                              hipStream_t stream) {
  const float* x      = (const float*)d_in[0];
  const float* A      = (const float*)d_in[1];
  const float* Wqkv   = (const float*)d_in[2];
  const float* Wo     = (const float*)d_in[3];
  const float* bo     = (const float*)d_in[4];
  const float* ln1_g  = (const float*)d_in[5];
  const float* ln1_b  = (const float*)d_in[6];
  const float* Wg     = (const float*)d_in[7];
  const float* bg     = (const float*)d_in[8];
  const float* Wgcn   = (const float*)d_in[9];
  const float* bgcn   = (const float*)d_in[10];
  const float* conv_w = (const float*)d_in[11];
  const float* conv_b = (const float*)d_in[12];
  const float* ln2_g  = (const float*)d_in[13];
  const float* ln2_b  = (const float*)d_in[14];
  float* out = (float*)d_out;

  char* ws = (char*)d_ws;
  size_t off = 0;
  auto alloc = [&](size_t bytes) -> void* {
    void* p = (void*)(ws + off);
    off += (bytes + 255) & ~(size_t)255;
    return p;
  };
  __bf16* xb    = (__bf16*)alloc((size_t)NT * DIM * 2);
  __bf16* wqkvb = (__bf16*)alloc((size_t)DIM * TH3 * 2);
  __bf16* wob   = (__bf16*)alloc((size_t)INNER * DIM * 2);
  __bf16* qkvb  = (__bf16*)alloc((size_t)NT * TH3 * 2);
  __bf16* attnb = (__bf16*)alloc((size_t)NT * INNER * 2);
  float*  oproj = (float*)alloc((size_t)NT * DIM * 4);
  float*  ybuf  = (float*)alloc((size_t)NT * DIM * 4);
  float*  yout  = (float*)alloc((size_t)NT * DIM * 4);
  float*  yconv = (float*)alloc((size_t)NT * DIM * 4);

  // 1) bf16 conversions
  cvt_bf16_kernel<<<(NT * DIM + 255) / 256, 256, 0, stream>>>(x, xb, NT * DIM);
  cvt_bf16_kernel<<<(DIM * TH3 + 255) / 256, 256, 0, stream>>>(Wqkv, wqkvb, DIM * TH3);
  cvt_bf16_kernel<<<(INNER * DIM + 255) / 256, 256, 0, stream>>>(Wo, wob, INNER * DIM);

  // 2) QKV projection: [8192,320] @ [320,1536] -> bf16
  gemm_bf16_kernel<0><<<dim3(NT / 64, TH3 / 32), 256, 0, stream>>>(
      xb, wqkvb, nullptr, qkvb, NT, TH3, DIM);

  // 3) fused attention (208000 B dynamic LDS; 320 KB/WGP on CDNA5)
  attention_kernel<<<dim3(TT / 16, HH, NB), 256, 208000, stream>>>(qkvb, attnb);

  // 4) output projection: [8192,512] @ [512,320] -> f32
  gemm_bf16_kernel<1><<<dim3(NT / 64, DIM / 32), 256, 0, stream>>>(
      attnb, wob, oproj, nullptr, NT, DIM, INNER);

  // 5) +bo, LayerNorm, LeakyReLU, +x
  ln1_kernel<<<NT, 320, 0, stream>>>(oproj, bo, ln1_g, ln1_b, x, ybuf);

  // 6) MoE gate + masked expert GCN + residual
  moe_gcn_kernel<<<NT, 320, 0, stream>>>(ybuf, A, Wg, bg, Wgcn, bgcn, yout);

  // 7) causal conv over time
  conv_kernel<<<(NT * DIM + 255) / 256, 256, 0, stream>>>(yout, conv_w, conv_b, yconv);

  // 8) final LayerNorm + LeakyReLU -> out
  ln2_kernel<<<NT, 320, 0, stream>>>(yconv, ln2_g, ln2_b, out);
}
